// MoELayer_30459908063733
// MI455X (gfx1250) — compile-verified, matrix operand order checked by reading, not run on hardware
//
#include <hip/hip_runtime.h>
#include <hip/hip_bf16.h>

// ---------------- problem constants ----------------
#define T_TOKENS 16384   // B*S = 4*4096
#define DD 512           // model dim
#define FF 1024          // ffn dim
#define EE 8             // experts
#define MT 64            // token tile (rows) per FFN block
#define BPAD 40          // padded B-tile row stride in halves (32 + 8 pad)

typedef __bf16 bf16_t;
typedef bf16_t v16bf __attribute__((ext_vector_type(16)));
typedef float  v8f   __attribute__((ext_vector_type(8)));

union FragU { v16bf v; uint4 q[2]; };

// ---- CDNA5 async global->LDS DMA (ASYNCcnt), with sync fallback ----
#if defined(__has_builtin)
#if __has_builtin(__builtin_amdgcn_global_load_async_to_lds_b128) && \
    __has_builtin(__builtin_amdgcn_s_wait_asynccnt)
#define HAVE_ASYNC_LDS 1
#endif
#endif

// Builtin expects: (int4 addrspace(1)*, int4 addrspace(3)*, imm offset, imm cpol)
typedef int v4i_gcc __attribute__((vector_size(16)));
typedef __attribute__((address_space(1))) v4i_gcc as1_v4i;
typedef __attribute__((address_space(3))) v4i_gcc as3_v4i;

__device__ __forceinline__ void async_copy16(const bf16_t* __restrict__ g,
                                             bf16_t* __restrict__ l) {
#ifdef HAVE_ASYNC_LDS
    __builtin_amdgcn_global_load_async_to_lds_b128(
        (as1_v4i*)(uintptr_t)g, (as3_v4i*)(uintptr_t)l, 0, 0);
#else
    *reinterpret_cast<uint4*>(l) = *reinterpret_cast<const uint4*>(g);
#endif
}

__device__ __forceinline__ void wait_async_lds() {
#ifdef HAVE_ASYNC_LDS
    __builtin_amdgcn_s_wait_asynccnt(0);
#endif
}

// Load one WMMA 16x16x32 bf16 A/B fragment for this lane.
// Layout (ISA 7.12.2, 16-bit A 16x32): lanes 0-15 hold row M=lane with
// K = {kc..kc+7, kc+16..kc+23}; lanes 16-31 hold row M=lane-16 with
// K = {kc+8..kc+15, kc+24..kc+31}. B (32x16) mirrors with lanes = N.
__device__ __forceinline__ v16bf load_frag(const bf16_t* __restrict__ rowbase,
                                           int kc, int hi) {
    FragU f;
    f.q[0] = *reinterpret_cast<const uint4*>(rowbase + kc + hi * 8);
    f.q[1] = *reinterpret_cast<const uint4*>(rowbase + kc + 16 + hi * 8);
    return f.v;
}

// Stage a 512-col x 32-k B tile into LDS (one row per thread, 4x16B each).
__device__ __forceinline__ void stage_tile(const bf16_t* __restrict__ src_base,
                                           int stride_halves, int col0, int kc,
                                           bf16_t* __restrict__ dst) {
    const int r = threadIdx.x;   // 512 threads == 512 B-rows
    const bf16_t* src = src_base + (size_t)(col0 + r) * stride_halves + kc;
    bf16_t* d = dst + r * BPAD;
    #pragma unroll
    for (int c = 0; c < 4; ++c)
        async_copy16(src + c * 8, d + c * 8);
}

// ---------------- small prep kernels ----------------
__global__ void zero_cnt_kernel(int* __restrict__ cnt) {
    if (threadIdx.x < EE) cnt[threadIdx.x] = 0;
}

// w1: [E][D][F] f32  ->  w1t: [E][F][D] bf16   (B of GEMM1, K=D contiguous)
__global__ void prep_w1_kernel(const float* __restrict__ w1, bf16_t* __restrict__ w1t) {
    int idx = blockIdx.x * blockDim.x + threadIdx.x;
    if (idx >= EE * FF * DD) return;
    int d = idx % DD;
    int f = (idx / DD) % FF;
    int e = idx / (DD * FF);
    w1t[idx] = (bf16_t)w1[(size_t)e * DD * FF + (size_t)d * FF + f];
}

// w2: [E][F][D] f32  ->  w2t: [E][D][F] bf16   (B of GEMM2, K=F contiguous)
__global__ void prep_w2_kernel(const float* __restrict__ w2, bf16_t* __restrict__ w2t) {
    int idx = blockIdx.x * blockDim.x + threadIdx.x;
    if (idx >= EE * DD * FF) return;
    int f = idx % FF;
    int d = (idx / FF) % DD;
    int e = idx / (FF * DD);
    w2t[idx] = (bf16_t)w2[(size_t)e * FF * DD + (size_t)f * DD + d];
}

// ---------------- gating: logits -> top2 -> softmax -> routing lists ----------------
__global__ __launch_bounds__(128)
void gate_kernel(const float* __restrict__ x, const float* __restrict__ wg,
                 const float* __restrict__ bg,
                 bf16_t* __restrict__ xb, float* __restrict__ out,
                 int* __restrict__ cnt, int* __restrict__ ids,
                 float* __restrict__ gts) {
    __shared__ float swg[DD * EE];   // 16 KB gating matrix
    for (int i = threadIdx.x; i < DD * EE; i += blockDim.x) swg[i] = wg[i];
    __syncthreads();

    int t = blockIdx.x * blockDim.x + threadIdx.x;
    if (t >= T_TOKENS) return;

    const float* xr  = x  + (size_t)t * DD;
    bf16_t*     xbr  = xb + (size_t)t * DD;
    float*      outr = out + (size_t)t * DD;

    float acc[EE];
    #pragma unroll
    for (int e = 0; e < EE; ++e) acc[e] = bg[e];

    for (int d = 0; d < DD; ++d) {
        float xv = xr[d];
        xbr[d] = (bf16_t)xv;     // x -> bf16 for WMMA
        outr[d] = 0.0f;          // zero-init output row (accumulated later via atomics)
        #pragma unroll
        for (int e = 0; e < EE; ++e) acc[e] += xv * swg[d * EE + e];
    }

    // top-2 (first index wins ties, like jax.lax.top_k)
    int   i0 = 0;
    float v0 = acc[0];
    #pragma unroll
    for (int e = 1; e < EE; ++e) if (acc[e] > v0) { v0 = acc[e]; i0 = e; }
    int   i1 = (i0 == 0) ? 1 : 0;
    float v1 = acc[i1];
    #pragma unroll
    for (int e = 0; e < EE; ++e)
        if (e != i0 && acc[e] > v1) { v1 = acc[e]; i1 = e; }

    // softmax over the two selected logits (v0 >= v1)
    float g0 = 1.0f / (1.0f + __expf(v1 - v0));
    float g1 = 1.0f - g0;

    int s0 = atomicAdd(&cnt[i0], 1);
    ids[i0 * T_TOKENS + s0] = t;
    gts[i0 * T_TOKENS + s0] = g0;
    int s1 = atomicAdd(&cnt[i1], 1);
    ids[i1 * T_TOKENS + s1] = t;
    gts[i1 * T_TOKENS + s1] = g1;
}

// ---------------- fused expert FFN: y = relu(Xg W1 + b1) W2 + b2, scatter*gate ----------------
__global__ __launch_bounds__(512)
void ffn_kernel(const bf16_t* __restrict__ xb,
                const bf16_t* __restrict__ w1t,   // [E][F][D]
                const bf16_t* __restrict__ w2t,   // [E][D][F]
                const float* __restrict__ b1,     // [E][F]
                const float* __restrict__ b2,     // [E][D]
                const int* __restrict__ cnt,
                const int* __restrict__ ids,
                const float* __restrict__ gts,
                float* __restrict__ out) {
    __shared__ __align__(16) bf16_t s_h[MT][FF];        // 128 KB h-tile
    __shared__ __align__(16) bf16_t s_b[2][512][BPAD];  // 80 KB double-buffered B tile
    __shared__ int   s_tok[MT];
    __shared__ float s_gate[MT];

    const int e     = blockIdx.x;
    const int count = cnt[e];
    const int m0    = blockIdx.y * MT;
    if (m0 >= count) return;   // uniform across block -> EXEC stays all-ones

    for (int i = threadIdx.x; i < MT; i += 512) {
        int r = m0 + i;
        if (r < count) { s_tok[i] = ids[e * T_TOKENS + r]; s_gate[i] = gts[e * T_TOKENS + r]; }
        else           { s_tok[i] = 0;                     s_gate[i] = 0.0f; }
    }
    __syncthreads();

    const int wave = threadIdx.x >> 5;
    const int lane = threadIdx.x & 31;
    const int l    = lane & 15;
    const int hi   = lane >> 4;

    const bf16_t* w1e = w1t + (size_t)e * FF * DD;
    const bf16_t* w2e = w2t + (size_t)e * DD * FF;

    const int mg = wave & 3;   // 4 row-groups of 16
    const int ng = wave >> 2;  // 4 col-groups of 128

    // gathered A row for this lane (row M = mg*16 + l of the tile)
    const bf16_t* arow = xb + (size_t)s_tok[mg * 16 + l] * DD;

    // ---------- Phase 1: h = relu(Xg @ W1 + b1)  (N = 1024 in two passes of 512) ----------
    for (int p = 0; p < 2; ++p) {
        const int cbase = p * 512;
        v8f acc[8];
        #pragma unroll
        for (int nb = 0; nb < 8; ++nb)
            #pragma unroll
            for (int j = 0; j < 8; ++j) acc[nb][j] = 0.0f;

        stage_tile(w1e, DD, cbase, 0, &s_b[0][0][0]);   // prologue: chunk 0 -> buf 0
        int buf = 0;
        for (int kci = 0; kci < DD / 32; ++kci) {
            const int kc = kci * 32;
            wait_async_lds();          // this wave's DMA done
            __syncthreads();           // -> everyone's DMA visible
            if (kci + 1 < DD / 32)     // prefetch next chunk into other buffer
                stage_tile(w1e, DD, cbase, kc + 32, &s_b[buf ^ 1][0][0]);

            v16bf a = load_frag(arow, kc, hi);
            #pragma unroll
            for (int nb = 0; nb < 8; ++nb) {
                v16bf b = load_frag(&s_b[buf][ng * 128 + nb * 16 + l][0], 0, hi);
                acc[nb] = __builtin_amdgcn_wmma_f32_16x16x32_bf16(
                    false, a, false, b, (short)0, acc[nb], false, false);
            }
            __syncthreads();           // all reads of buf done before it is re-staged
            buf ^= 1;
        }
        // epilogue: +b1, relu, bf16 -> LDS h-tile
        #pragma unroll
        for (int nb = 0; nb < 8; ++nb) {
            const int col  = cbase + ng * 128 + nb * 16 + l;
            const float bv = b1[e * FF + col];
            #pragma unroll
            for (int j = 0; j < 8; ++j) {
                const int row = mg * 16 + j + 8 * hi;
                float v = acc[nb][j] + bv;
                v = v > 0.0f ? v : 0.0f;
                s_h[row][col] = (bf16_t)v;
            }
        }
    }
    __syncthreads();

    // ---------- Phase 2: y = h @ W2 + b2 ; out[tok] += gate * y ----------
    {
        v8f acc[8];
        #pragma unroll
        for (int nb = 0; nb < 8; ++nb)
            #pragma unroll
            for (int j = 0; j < 8; ++j) acc[nb][j] = 0.0f;

        const bf16_t* hrow = &s_h[mg * 16 + l][0];
        stage_tile(w2e, FF, 0, 0, &s_b[0][0][0]);
        int buf = 0;
        for (int kci = 0; kci < FF / 32; ++kci) {
            const int kc = kci * 32;
            wait_async_lds();
            __syncthreads();
            if (kci + 1 < FF / 32)
                stage_tile(w2e, FF, 0, kc + 32, &s_b[buf ^ 1][0][0]);

            v16bf a = load_frag(hrow, kc, hi);   // ds_load_b128 pairs from LDS h
            #pragma unroll
            for (int nb = 0; nb < 8; ++nb) {
                v16bf b = load_frag(&s_b[buf][ng * 128 + nb * 16 + l][0], 0, hi);
                acc[nb] = __builtin_amdgcn_wmma_f32_16x16x32_bf16(
                    false, a, false, b, (short)0, acc[nb], false, false);
            }
            __syncthreads();
            buf ^= 1;
        }
        #pragma unroll
        for (int nb = 0; nb < 8; ++nb) {
            const int col  = ng * 128 + nb * 16 + l;
            const float bv = b2[e * DD + col];
            #pragma unroll
            for (int j = 0; j < 8; ++j) {
                const int row = mg * 16 + j + 8 * hi;
                const float g = s_gate[row];
                if (g != 0.0f) {
                    const float v = (acc[nb][j] + bv) * g;
                    atomicAdd(&out[(size_t)s_tok[row] * DD + col], v);
                }
            }
        }
    }
}

// ---------------- host launcher ----------------
extern "C" void kernel_launch(void* const* d_in, const int* in_sizes, int n_in,
                              void* d_out, int out_size, void* d_ws, size_t ws_size,
                              hipStream_t stream) {
    const float* x  = (const float*)d_in[0];
    const float* w1 = (const float*)d_in[1];
    const float* b1 = (const float*)d_in[2];
    const float* w2 = (const float*)d_in[3];
    const float* b2 = (const float*)d_in[4];
    const float* wg = (const float*)d_in[5];
    const float* bg = (const float*)d_in[6];
    float* out = (float*)d_out;

    char* ws = (char*)d_ws;
    size_t off = 0;
    bf16_t* xb  = (bf16_t*)(ws + off); off += (size_t)T_TOKENS * DD * sizeof(bf16_t);
    bf16_t* w1t = (bf16_t*)(ws + off); off += (size_t)EE * FF * DD * sizeof(bf16_t);
    bf16_t* w2t = (bf16_t*)(ws + off); off += (size_t)EE * DD * FF * sizeof(bf16_t);
    int*    cnt = (int*)(ws + off);    off += 64;
    int*    ids = (int*)(ws + off);    off += (size_t)EE * T_TOKENS * sizeof(int);
    float*  gts = (float*)(ws + off);  off += (size_t)EE * T_TOKENS * sizeof(float);
    (void)ws_size; (void)in_sizes; (void)n_in; (void)out_size;

    zero_cnt_kernel<<<1, 32, 0, stream>>>(cnt);

    const int wn = EE * FF * DD;
    prep_w1_kernel<<<(wn + 255) / 256, 256, 0, stream>>>(w1, w1t);
    prep_w2_kernel<<<(wn + 255) / 256, 256, 0, stream>>>(w2, w2t);

    gate_kernel<<<T_TOKENS / 128, 128, 0, stream>>>(x, wg, bg, xb, out, cnt, ids, gts);

    dim3 grid(EE, (T_TOKENS + MT - 1) / MT);   // blocks past a tile's count exit early
    ffn_kernel<<<grid, 512, 0, stream>>>(xb, w1t, w2t, b1, b2, cnt, ids, gts, out);
}
